// BoxHead_42133629174425
// MI455X (gfx1250) — matrix operand order repaired and checked by reading, not compiled, not run on hardware
//
#include <hip/hip_runtime.h>

typedef __bf16 bf16_t;
typedef __attribute__((ext_vector_type(16))) __bf16 v16bf;
typedef __attribute__((ext_vector_type(8)))  __bf16 v8bf;
typedef __attribute__((ext_vector_type(4)))  __bf16 v4bf;
typedef __attribute__((ext_vector_type(8)))  float  v8f;

#define N_PROP 16384
#define FEAT   12544
#define HID    1024
#define NCLS   4     // C+1
#define NREG   12    // 4*C
#define NHEAD  16

__device__ __forceinline__ bf16_t f2bf(float f) { return (bf16_t)f; }

union FragAB { v16bf v; v8bf h[2]; };

// LDS-relative byte address of a __shared__ object (generic -> addrspace(3))
__device__ __forceinline__ unsigned lds_addr(const void* p)
{
    return (unsigned)(unsigned long long)(__attribute__((address_space(3))) const void*)p;
}

// CDNA5 async global->LDS copy, 16B per lane, tracked by ASYNCcnt
__device__ __forceinline__ void async_cp_b128(unsigned lds_off, const void* gptr)
{
    asm volatile("global_load_async_to_lds_b128 %0, %1, off"
                 :: "v"(lds_off), "v"((unsigned long long)(size_t)gptr)
                 : "memory");
}

__device__ __forceinline__ void wait_asynccnt0()
{
    asm volatile("s_wait_asynccnt 0" ::: "memory");
}

// ---------------------------------------------------------------------------
// Transpose + f32->bf16 convert: W [K,N] row-major -> Wt [N,K] row-major.
// K, N multiples of 32.
// ---------------------------------------------------------------------------
__global__ __launch_bounds__(256)
void k_transpose_cvt(const float* __restrict__ W, bf16_t* __restrict__ Wt,
                     int K, int N)
{
    __shared__ bf16_t tile[32][33];
    const int n0 = blockIdx.x * 32;
    const int k0 = blockIdx.y * 32;
    const int tx = threadIdx.x & 31;
    const int ty = threadIdx.x >> 5;  // 0..7
#pragma unroll
    for (int i = 0; i < 32; i += 8)
        tile[ty + i][tx] = f2bf(W[(size_t)(k0 + ty + i) * N + (n0 + tx)]);
    __syncthreads();
#pragma unroll
    for (int i = 0; i < 32; i += 8)
        Wt[(size_t)(n0 + ty + i) * K + (k0 + tx)] = tile[tx][ty + i];
}

// ---------------------------------------------------------------------------
// Pack Wc [1024,4] and Wr [1024,12] into WcrT bf16 [16][1024]
// ---------------------------------------------------------------------------
__global__ __launch_bounds__(256)
void k_build_headT(const float* __restrict__ Wc, const float* __restrict__ Wr,
                   bf16_t* __restrict__ WcrT)
{
    int idx = blockIdx.x * blockDim.x + threadIdx.x;
    if (idx >= NHEAD * HID) return;
    int k = idx & (HID - 1);
    int n = idx >> 10;
    float v = (n < NCLS) ? Wc[(size_t)k * NCLS + n]
                         : Wr[(size_t)k * NREG + (n - NCLS)];
    WcrT[(size_t)n * HID + k] = f2bf(v);
}

// ---------------------------------------------------------------------------
// Fused GEMM + bias + ReLU, bf16 WMMA, fp32 accumulate.
//   A [M,K] (f32 converted on the fly, or bf16 via async->LDS),
//   Bt [N,K] bf16 (async->LDS), Out [M,N] bf16.
//   Block tile 128x128, BK=32, 8 waves, double-buffered LDS.
// ---------------------------------------------------------------------------
template <bool A_F32>
__global__ __launch_bounds__(256)
void k_gemm_relu(const void* __restrict__ Aptr,
                 const bf16_t* __restrict__ Bt,
                 const float* __restrict__ bias,
                 bf16_t* __restrict__ Out,
                 int M, int N, int K)
{
    constexpr int BM = 128, BN = 128, BK = 32;
    __shared__ __align__(64) bf16_t As[2][BM][BK];   // 2 * 8 KB
    __shared__ __align__(64) bf16_t Bs[2][BN][BK];   // 2 * 8 KB

    const int tid   = threadIdx.x;
    const int wid   = tid >> 5;
    const int lane  = tid & 31;
    const int l16   = lane & 15;
    const int khalf = lane >> 4;

    const int wm = wid >> 1;   // 0..3 -> 32-row slice of the block tile
    const int wn = wid & 1;    // 0..1 -> 64-col slice

    const int blockN = blockIdx.x * BN;
    const int blockM = blockIdx.y * BM;

    const v8f vzero = {0.f, 0.f, 0.f, 0.f, 0.f, 0.f, 0.f, 0.f};
    v8f acc[2][4];
#pragma unroll
    for (int i = 0; i < 2; ++i)
#pragma unroll
        for (int j = 0; j < 4; ++j) acc[i][j] = vzero;

    // loader thread mapping
    const int ar  = tid >> 3;        // f32 A: 0..31, 4 row-passes of 32
    const int ac4 = (tid & 7) * 4;
    const int hr  = tid >> 2;        // bf16 tiles: 0..63, 2 row-passes of 64
    const int hc8 = (tid & 3) * 8;

    const int nk = K / BK;

    // ---- prologue: stage tile 0 into buffer 0 ----
    if constexpr (A_F32) {
        const float* A = (const float*)Aptr;
#pragma unroll
        for (int i = 0; i < 4; ++i) {
            int r = ar + i * 32;
            float4 t = *(const float4*)(A + (size_t)(blockM + r) * K + ac4);
            v4bf p; p[0] = f2bf(t.x); p[1] = f2bf(t.y); p[2] = f2bf(t.z); p[3] = f2bf(t.w);
            *(v4bf*)&As[0][r][ac4] = p;
        }
    } else {
        const bf16_t* A = (const bf16_t*)Aptr;
#pragma unroll
        for (int i = 0; i < 2; ++i) {
            int r = hr + i * 64;
            async_cp_b128(lds_addr(&As[0][r][hc8]),
                          A + (size_t)(blockM + r) * K + hc8);
        }
    }
#pragma unroll
    for (int i = 0; i < 2; ++i) {
        int r = hr + i * 64;
        async_cp_b128(lds_addr(&Bs[0][r][hc8]),
                      Bt + (size_t)(blockN + r) * K + hc8);
    }
    wait_asynccnt0();
    __syncthreads();

    int buf = 0;
    for (int kt = 0; kt < nk; ++kt) {
        const bool has_next = (kt + 1) < nk;
        const int  nb = buf ^ 1;
        float4 aregs[4];

        // ---- prefetch next tile: async straight to LDS (bf16 operands),
        //      VGPR round-trip only for the f32->bf16 converted A path ----
        if (has_next) {
            const int k0 = (kt + 1) * BK;
            if constexpr (A_F32) {
                const float* A = (const float*)Aptr;
#pragma unroll
                for (int i = 0; i < 4; ++i) {
                    int r = ar + i * 32;
                    aregs[i] = *(const float4*)(A + (size_t)(blockM + r) * K + k0 + ac4);
                }
            } else {
                const bf16_t* A = (const bf16_t*)Aptr;
#pragma unroll
                for (int i = 0; i < 2; ++i) {
                    int r = hr + i * 64;
                    async_cp_b128(lds_addr(&As[nb][r][hc8]),
                                  A + (size_t)(blockM + r) * K + k0 + hc8);
                }
            }
#pragma unroll
            for (int i = 0; i < 2; ++i) {
                int r = hr + i * 64;
                async_cp_b128(lds_addr(&Bs[nb][r][hc8]),
                              Bt + (size_t)(blockN + r) * K + k0 + hc8);
            }
        }

        // ---- compute current tile from LDS (ISA fragment layouts) ----
        {
            FragAB a[2], b[4];
#pragma unroll
            for (int i = 0; i < 2; ++i) {
                int r = wm * 32 + i * 16 + l16;
                a[i].h[0] = *(const v8bf*)&As[buf][r][khalf * 8];        // K +0..+7
                a[i].h[1] = *(const v8bf*)&As[buf][r][khalf * 8 + 16];   // K +16..+23
            }
#pragma unroll
            for (int j = 0; j < 4; ++j) {
                int c = wn * 64 + j * 16 + l16;
                b[j].h[0] = *(const v8bf*)&Bs[buf][c][khalf * 8];        // K +0..+7
                b[j].h[1] = *(const v8bf*)&Bs[buf][c][khalf * 8 + 16];   // K +16..+23
            }
#pragma unroll
            for (int i = 0; i < 2; ++i)
#pragma unroll
                for (int j = 0; j < 4; ++j)
                    acc[i][j] = __builtin_amdgcn_wmma_f32_16x16x32_bf16(
                        false, a[i].v, false, b[j].v, (short)0, acc[i][j],
                        false, false);
        }

        // ---- commit converted f32 A tile, close the async stage ----
        if (has_next) {
            if constexpr (A_F32) {
#pragma unroll
                for (int i = 0; i < 4; ++i) {
                    int r = ar + i * 32;
                    v4bf p;
                    p[0] = f2bf(aregs[i].x); p[1] = f2bf(aregs[i].y);
                    p[2] = f2bf(aregs[i].z); p[3] = f2bf(aregs[i].w);
                    *(v4bf*)&As[nb][r][ac4] = p;
                }
            }
            wait_asynccnt0();
        }
        __syncthreads();
        buf = nb;
    }

    // ---- epilogue: bias + ReLU, store bf16 ----
#pragma unroll
    for (int i = 0; i < 2; ++i) {
#pragma unroll
        for (int j = 0; j < 4; ++j) {
            int gn = blockN + wn * 64 + j * 16 + l16;
            float bv = bias[gn];
            int rbase = blockM + wm * 32 + i * 16 + khalf * 8;
#pragma unroll
            for (int r = 0; r < 8; ++r) {
                float v = acc[i][j][r] + bv;
                v = v > 0.f ? v : 0.f;
                Out[(size_t)(rbase + r) * N + gn] = f2bf(v);
            }
        }
    }
}

// ---------------------------------------------------------------------------
// Head GEMM: H2 [M,1024] bf16 x WcrT [16,1024] bf16 -> logits/boxes (f32).
// One wave per 16 rows; output scattered into concatenated d_out.
// ---------------------------------------------------------------------------
__global__ __launch_bounds__(256)
void k_head(const bf16_t* __restrict__ H, const bf16_t* __restrict__ WcrT,
            const float* __restrict__ bc, const float* __restrict__ br,
            float* __restrict__ out, int M, int K)
{
    const int wid   = threadIdx.x >> 5;
    const int lane  = threadIdx.x & 31;
    const int l16   = lane & 15;
    const int khalf = lane >> 4;
    const int m0    = (blockIdx.x * 8 + wid) * 16;

    v8f acc = {0.f, 0.f, 0.f, 0.f, 0.f, 0.f, 0.f, 0.f};
    for (int k0 = 0; k0 < K; k0 += 32) {
        FragAB a, b;
        const bf16_t* arow = H + (size_t)(m0 + l16) * K + k0 + khalf * 8;
        a.h[0] = *(const v8bf*)arow;
        a.h[1] = *(const v8bf*)(arow + 16);
        const bf16_t* brow = WcrT + (size_t)l16 * K + k0 + khalf * 8;
        b.h[0] = *(const v8bf*)brow;
        b.h[1] = *(const v8bf*)(brow + 16);
        acc = __builtin_amdgcn_wmma_f32_16x16x32_bf16(
            false, a.v, false, b.v, (short)0, acc, false, false);
    }

    const int col = l16;
    float bias = (col < NCLS) ? bc[col] : br[col - NCLS];
#pragma unroll
    for (int r = 0; r < 8; ++r) {
        int row = m0 + r + khalf * 8;
        float v = acc[r] + bias;
        if (col < NCLS)
            out[(size_t)row * NCLS + col] = v;
        else
            out[(size_t)NCLS * M + (size_t)row * NREG + (col - NCLS)] = v;
    }
}

// ---------------------------------------------------------------------------
extern "C" void kernel_launch(void* const* d_in, const int* in_sizes, int n_in,
                              void* d_out, int out_size, void* d_ws, size_t ws_size,
                              hipStream_t stream)
{
    const float* X  = (const float*)d_in[0];
    const float* W1 = (const float*)d_in[1];
    const float* b1 = (const float*)d_in[2];
    const float* W2 = (const float*)d_in[3];
    const float* b2 = (const float*)d_in[4];
    const float* Wc = (const float*)d_in[5];
    const float* bc = (const float*)d_in[6];
    const float* Wr = (const float*)d_in[7];
    const float* br = (const float*)d_in[8];
    float* out = (float*)d_out;

    char*  ws  = (char*)d_ws;
    size_t off = 0;
    auto alloc = [&](size_t bytes) -> void* {
        void* p = ws + off;
        off = (off + bytes + 255) & ~(size_t)255;
        return p;
    };
    bf16_t* W1T  = (bf16_t*)alloc((size_t)FEAT * HID * sizeof(bf16_t));   // 25.7 MB
    bf16_t* W2T  = (bf16_t*)alloc((size_t)HID * HID * sizeof(bf16_t));    //  2.1 MB
    bf16_t* WcrT = (bf16_t*)alloc((size_t)NHEAD * HID * sizeof(bf16_t));  // 32 KB
    bf16_t* H1   = (bf16_t*)alloc((size_t)N_PROP * HID * sizeof(bf16_t)); // 33.6 MB
    bf16_t* H2   = (bf16_t*)alloc((size_t)N_PROP * HID * sizeof(bf16_t)); // 33.6 MB

    // one-time weight transpose+convert (tiny vs the 822 MB activation stream)
    k_transpose_cvt<<<dim3(HID / 32, FEAT / 32), 256, 0, stream>>>(W1, W1T, FEAT, HID);
    k_transpose_cvt<<<dim3(HID / 32, HID / 32), 256, 0, stream>>>(W2, W2T, HID, HID);
    k_build_headT<<<(NHEAD * HID + 255) / 256, 256, 0, stream>>>(Wc, Wr, WcrT);

    // GEMM1: X f32 -> H1 bf16 (bias+ReLU fused). Grid N-fastest for L2 reuse of X.
    k_gemm_relu<true><<<dim3(HID / 128, N_PROP / 128), 256, 0, stream>>>(
        (const void*)X, W1T, b1, H1, N_PROP, HID, FEAT);

    // GEMM2: H1 bf16 -> H2 bf16 (bias+ReLU fused)
    k_gemm_relu<false><<<dim3(HID / 128, N_PROP / 128), 256, 0, stream>>>(
        (const void*)H1, W2T, b2, H2, N_PROP, HID, HID);

    // heads: fused class+box projection straight into d_out
    k_head<<<N_PROP / 128, 256, 0, stream>>>(H2, WcrT, bc, br, out, N_PROP, HID);
}
